// DiTAttention_40303973105877
// MI455X (gfx1250) — matrix operand order repaired and checked by reading, not compiled, not run on hardware
//
#include <hip/hip_runtime.h>
#include <hip/hip_bf16.h>

// ---------------------------------------------------------------------------
// DiT attention block for MI455X (gfx1250, wave32, WMMA).
// Pipeline: f32->bf16 cvt | QKV GEMM (bf16 WMMA, f32 acc) | RMSNorm+2D-RoPE |
//           flash attention (bf16 WMMA, online softmax f32) | proj GEMM.
// Data movement: double-buffered GLOBAL_LOAD_ASYNC_TO_LDS_B128 (ASYNCcnt)
// when the toolchain exposes the builtins, else staged through VGPRs.
// ---------------------------------------------------------------------------

#ifndef __has_builtin
#define __has_builtin(x) 0
#endif

#if __has_builtin(__builtin_amdgcn_global_load_async_to_lds_b128) && \
    __has_builtin(__builtin_amdgcn_s_wait_asynccnt)
#define USE_ASYNC_LDS 1
#else
#define USE_ASYNC_LDS 0
#endif

typedef __bf16 bf16_t;
typedef __attribute__((ext_vector_type(16))) __bf16 v16bf;
typedef __attribute__((ext_vector_type(8)))  __bf16 v8bf;
typedef __attribute__((ext_vector_type(8)))  float  v8f;
typedef __attribute__((ext_vector_type(4)))  int    v4i;

__device__ __forceinline__ v8f wmma_bf16f32(v16bf a, v16bf b, v8f c) {
  // D = A(16x32 bf16) * B(32x16 bf16) + C(16x16 f32)
  return __builtin_amdgcn_wmma_f32_16x16x32_bf16(
      /*neg_a=*/false, a, /*neg_b=*/false, b,
      /*c_mod=*/(short)0, c, /*reuse_a=*/false, /*reuse_b=*/false);
}

__device__ __forceinline__ v16bf cat8(v8bf lo, v8bf hi) {
  return __builtin_shufflevector(lo, hi, 0, 1, 2, 3, 4, 5, 6, 7,
                                 8, 9, 10, 11, 12, 13, 14, 15);
}

// Copy 16B global -> LDS. Async path goes straight to LDS (no VGPR bounce).
// Builtin signature (from the toolchain diagnostic): v4i AS1* src,
// v4i AS3* lds, imm offset, imm cpol.
__device__ __forceinline__ void cp16_g2l(bf16_t* lds, const bf16_t* g) {
#if USE_ASYNC_LDS
  __builtin_amdgcn_global_load_async_to_lds_b128(
      (__attribute__((address_space(1))) v4i*)(g),
      (__attribute__((address_space(3))) v4i*)(lds), 0, 0);
#else
  *(v8bf*)lds = *(const v8bf*)g;
#endif
}

__device__ __forceinline__ void async_wait0() {
#if USE_ASYNC_LDS
  __builtin_amdgcn_s_wait_asynccnt(0);
#endif
}

// A fragment, 16x32 bf16, row-major source with row stride `stride`.
// lane<16: row=lane,    K = {kBase+0..7,  kBase+16..23}
// lane>=16: row=lane-16, K = {kBase+8..15, kBase+24..31}
__device__ __forceinline__ v16bf load_a(const bf16_t* base, int stride,
                                        int kBase, int lane) {
  int m   = lane & 15;
  int klo = kBase + ((lane & 16) ? 8 : 0);
  const bf16_t* p = base + (size_t)m * stride + klo;
  return cat8(*(const v8bf*)p, *(const v8bf*)(p + 16));
}

// B fragment, 32x16 bf16, from [N][K] row-major storage (B^T contraction).
// lane<16: col=lane,    K = {kBase+0..7,  kBase+16..23}
// lane>=16: col=lane-16, K = {kBase+8..15, kBase+24..31}
__device__ __forceinline__ v16bf load_b(const bf16_t* base, int stride,
                                        int nBase, int kBase, int lane) {
  int n  = nBase + (lane & 15);
  int k0 = kBase + ((lane & 16) ? 8 : 0);
  const bf16_t* p = base + (size_t)n * stride + k0;
  return cat8(*(const v8bf*)p, *(const v8bf*)(p + 16));
}

// ---------------------------------------------------------------------------
// f32 -> bf16 conversion
// ---------------------------------------------------------------------------
__global__ void f32_to_bf16_kernel(const float* __restrict__ src,
                                   bf16_t* __restrict__ dst, int n) {
  int i = blockIdx.x * blockDim.x + threadIdx.x;
  int stride = gridDim.x * blockDim.x;
  for (; i < n; i += stride) dst[i] = (bf16_t)src[i];
}

// ---------------------------------------------------------------------------
// GEMM: C[M][N] = A[M][K] * B[N][K]^T + bias[N]   (A,B bf16; C f32)
// 128x128 block, 8 waves (4Mx2N), wave tile 32x64 (2x4 WMMA frags), K-step 32.
// Double-buffered async LDS staging: one barrier per K-step, DMA under WMMA.
// ---------------------------------------------------------------------------
#define GLDT 40  // padded LDS row stride (bf16 elems) to spread banks

__global__ __launch_bounds__(256) void gemm_bf16_nt(
    const bf16_t* __restrict__ A, const bf16_t* __restrict__ B,
    const float* __restrict__ bias, float* __restrict__ C,
    int M, int N, int K) {
  __shared__ bf16_t As[2][128 * GLDT];
  __shared__ bf16_t Bs[2][128 * GLDT];

  const int tid  = threadIdx.x;
  const int lane = tid & 31;
  const int wave = tid >> 5;
  const int wm = wave & 3;   // 0..3 : M sub-block
  const int wn = wave >> 2;  // 0..1 : N sub-block
  const int bm = blockIdx.y * 128;
  const int bn = blockIdx.x * 128;

  // cooperative stage: 256 threads * 32B = one 128x32 bf16 tile per matrix
  const int lrow = tid >> 1;
  const int lseg = (tid & 1) * 16;
  const bf16_t* gA = A + (size_t)(bm + lrow) * K + lseg;
  const bf16_t* gB = B + (size_t)(bn + lrow) * K + lseg;
  const int soff = lrow * GLDT + lseg;

  v8f acc[2][4] = {};

  // prologue: stage tile 0 into buffer 0
  cp16_g2l(&As[0][soff], gA);
  cp16_g2l(&As[0][soff + 8], gA + 8);
  cp16_g2l(&Bs[0][soff], gB);
  cp16_g2l(&Bs[0][soff + 8], gB + 8);
  async_wait0();
  __syncthreads();

  const int nk = K >> 5;
  for (int i = 0; i < nk; ++i) {
    const int cur = i & 1, nxt = cur ^ 1;
    if (i + 1 < nk) {  // kick off DMA for the next K-slab
      const int kb = (i + 1) << 5;
      cp16_g2l(&As[nxt][soff], gA + kb);
      cp16_g2l(&As[nxt][soff + 8], gA + kb + 8);
      cp16_g2l(&Bs[nxt][soff], gB + kb);
      cp16_g2l(&Bs[nxt][soff + 8], gB + kb + 8);
    }

    v16bf af0 = load_a(&As[cur][(wm * 32 + 0) * GLDT], GLDT, 0, lane);
    v16bf af1 = load_a(&As[cur][(wm * 32 + 16) * GLDT], GLDT, 0, lane);
#pragma unroll
    for (int nt = 0; nt < 4; ++nt) {
      v16bf bfr = load_b(Bs[cur], GLDT, wn * 64 + nt * 16, 0, lane);
      acc[0][nt] = wmma_bf16f32(af0, bfr, acc[0][nt]);
      acc[1][nt] = wmma_bf16f32(af1, bfr, acc[1][nt]);
    }

    async_wait0();   // next-tile DMA done (ds reads already consumed by WMMA)
    __syncthreads();
  }

  // epilogue: C/D layout col = lane%16, row = r + 8*(lane>=16)
  const int mhalf = (lane & 16) ? 8 : 0;
  const int ncol  = lane & 15;
#pragma unroll
  for (int mt = 0; mt < 2; ++mt)
#pragma unroll
    for (int nt = 0; nt < 4; ++nt) {
      int col  = bn + wn * 64 + nt * 16 + ncol;
      float bv = bias[col];
#pragma unroll
      for (int r = 0; r < 8; ++r) {
        int row = bm + wm * 32 + mt * 16 + r + mhalf;
        C[(size_t)row * N + col] = acc[mt][nt][r] + bv;
      }
    }
}

// ---------------------------------------------------------------------------
// RMSNorm (full 1024-dim) + axial 2D RoPE; emits Q/K/V bf16 [b][h][n][64].
// One workgroup (256 threads) per token.
// ---------------------------------------------------------------------------
__global__ __launch_bounds__(256) void norm_rope_kernel(
    const float* __restrict__ qkv, const float* __restrict__ q_scale,
    const float* __restrict__ k_scale, bf16_t* __restrict__ Qo,
    bf16_t* __restrict__ Ko, bf16_t* __restrict__ Vo,
    const int* __restrict__ widthp) {
  const int S = 2304, H = 16, HD = 64;
  __shared__ float redq[256];
  __shared__ float redk[256];
  const int t = blockIdx.x, b = blockIdx.y, tid = threadIdx.x;
  const float* row = qkv + ((size_t)b * S + t) * 3072;

  float sq = 0.f, sk = 0.f;
#pragma unroll
  for (int i = 0; i < 4; ++i) {
    int d = tid + i * 256;
    float xq = row[d], xk = row[1024 + d];
    sq += xq * xq;
    sk += xk * xk;
  }
  redq[tid] = sq; redk[tid] = sk;
  __syncthreads();
  for (int s = 128; s > 0; s >>= 1) {
    if (tid < s) { redq[tid] += redq[tid + s]; redk[tid] += redk[tid + s]; }
    __syncthreads();
  }
  const float rq = rsqrtf(redq[0] * (1.f / 1024.f) + 1e-6f);
  const float rk = rsqrtf(redk[0] * (1.f / 1024.f) + 1e-6f);

  const int w = *widthp;
  const float rowp = (float)(t / w);
  const float colp = (float)(t % w);
  const float kfreq = -13.287712379549449f / 16.f;  // -log2(10000)/16

#pragma unroll
  for (int i = 0; i < 4; ++i) {
    int d  = tid + i * 256;      // 0..1023
    int h  = d >> 6;
    int dd = d & 63;             // dim within head
    int ddl = dd & 31;           // dim within axial half (row / col keyed)
    float pos  = (dd < 32) ? rowp : colp;
    int   fi   = ddl & 15;
    float invf = exp2f((float)fi * kfreq);  // BASE^(-2*fi/32)
    float ang  = pos * invf;
    float c = cosf(ang), s = sinf(ang);
    int   dpart = d + ((ddl < 16) ? 16 : -16);  // rotate_half partner
    float sign  = (ddl < 16) ? -1.f : 1.f;

    float xq  = row[d] * rq * q_scale[d];
    float xqp = row[dpart] * rq * q_scale[dpart];
    float xk  = row[1024 + d] * rk * k_scale[d];
    float xkp = row[1024 + dpart] * rk * k_scale[dpart];

    size_t oidx = (((size_t)(b * H + h)) * S + t) * HD + dd;
    Qo[oidx] = (bf16_t)(xq * c + sign * xqp * s);
    Ko[oidx] = (bf16_t)(xk * c + sign * xkp * s);
    Vo[oidx] = (bf16_t)row[2048 + d];
  }
}

// ---------------------------------------------------------------------------
// Flash attention: grid (S/128, B*H). 8 waves, each owns 16 query rows.
// Key loop in tiles of 64, double-buffered (K async->LDS, V via VGPR
// transpose); online softmax f32; QK^T and PV on WMMA.
// ---------------------------------------------------------------------------
#define ALK 72  // padded LDS row stride for attention tiles

__global__ __launch_bounds__(256) void attn_fwd_kernel(
    const bf16_t* __restrict__ Q, const bf16_t* __restrict__ Kg,
    const bf16_t* __restrict__ Vg, bf16_t* __restrict__ O) {
  const int S = 2304, HD = 64, DIM = 1024;
  __shared__ bf16_t Ks[2][64 * ALK];      // key tile, [key][d]
  __shared__ bf16_t Vt[2][64 * ALK];      // value tile, transposed [d][key]
  __shared__ bf16_t Ps[8 * 16 * ALK];     // per-wave P scratch [16][64]

  const int tid  = threadIdx.x;
  const int lane = tid & 31;
  const int wave = tid >> 5;
  const int bh = blockIdx.y;
  const int b = bh >> 4, h = bh & 15;
  const int qBase = blockIdx.x * 128;

  const int lr = tid >> 2;        // 0..63 : tile row
  const int ls = (tid & 3) * 16;  // 0,16,32,48 : segment
  const bf16_t* gK = Kg + (size_t)bh * S * HD;
  const bf16_t* gV = Vg + (size_t)bh * S * HD;

  // prologue: stage key/value tile 0 into buffer 0
  cp16_g2l(&Ks[0][lr * ALK + ls], gK + (size_t)lr * HD + ls);
  cp16_g2l(&Ks[0][lr * ALK + ls + 8], gK + (size_t)lr * HD + ls + 8);
  {
    const bf16_t* vp = gV + (size_t)lr * HD + ls;
    v8bf v0 = *(const v8bf*)vp;
    v8bf v1 = *(const v8bf*)(vp + 8);
#pragma unroll
    for (int j = 0; j < 8; ++j) {
      Vt[0][(ls + j) * ALK + lr]     = v0[j];
      Vt[0][(ls + 8 + j) * ALK + lr] = v1[j];
    }
  }

  // this wave's Q fragments (16 rows x 64 dims), pre-scaled by HEAD_DIM^-0.5
  // (0.125 is a power of two: exact in bf16, commutes through the WMMA)
  const bf16_t* Qp = Q + ((size_t)bh * S + qBase + wave * 16) * HD;
  v16bf qf0 = load_a(Qp, HD, 0, lane);
  v16bf qf1 = load_a(Qp, HD, 32, lane);
#pragma unroll
  for (int e = 0; e < 16; ++e) {
    qf0[e] = (bf16_t)((float)qf0[e] * 0.125f);
    qf1[e] = (bf16_t)((float)qf1[e] * 0.125f);
  }

  v8f acc[4] = {};
  float mrow[8], lrow[8];
#pragma unroll
  for (int r = 0; r < 8; ++r) { mrow[r] = -1e30f; lrow[r] = 0.f; }

  bf16_t* Pw = &Ps[wave * 16 * ALK];
  const int mhalf = (lane & 16) ? 8 : 0;
  const int ncol  = lane & 15;

  async_wait0();
  __syncthreads();

  const int nkt = S / 64;
  for (int i = 0; i < nkt; ++i) {
    const int cur = i & 1, nxt = cur ^ 1;

    // kick off next tile: K straight to LDS (async), V into VGPRs now,
    // transposed store deferred under the QK^T compute.
    v8bf v0n = {}, v1n = {};
    const bool have_next = (i + 1 < nkt);
    if (have_next) {
      const size_t nb = (size_t)((i + 1) * 64 + lr) * HD + ls;
      cp16_g2l(&Ks[nxt][lr * ALK + ls], gK + nb);
      cp16_g2l(&Ks[nxt][lr * ALK + ls + 8], gK + nb + 8);
      v0n = *(const v8bf*)(gV + nb);
      v1n = *(const v8bf*)(gV + nb + 8);
    }

    // scores: S = (Q/8) * K^T  (4 key subtiles x 2 d-chunks of 32)
    v8f sf[4];
#pragma unroll
    for (int nt = 0; nt < 4; ++nt) {
      v8f s = {};
      s = wmma_bf16f32(qf0, load_b(Ks[cur], ALK, nt * 16, 0, lane), s);
      s = wmma_bf16f32(qf1, load_b(Ks[cur], ALK, nt * 16, 32, lane), s);
      sf[nt] = s;
    }

    if (have_next) {  // V transpose store overlaps the softmax below
#pragma unroll
      for (int j = 0; j < 8; ++j) {
        Vt[nxt][(ls + j) * ALK + lr]     = v0n[j];
        Vt[nxt][(ls + 8 + j) * ALK + lr] = v1n[j];
      }
    }

    // online softmax: row reductions stay within 16-lane halves (wave32)
    float mnew[8];
#pragma unroll
    for (int r = 0; r < 8; ++r) {
      float mx = fmaxf(fmaxf(sf[0][r], sf[1][r]), fmaxf(sf[2][r], sf[3][r]));
#pragma unroll
      for (int off = 8; off >= 1; off >>= 1)
        mx = fmaxf(mx, __shfl_xor(mx, off, 32));
      mnew[r] = fmaxf(mrow[r], mx);
      float alpha = expf(mrow[r] - mnew[r]);
      mrow[r] = mnew[r];
      lrow[r] *= alpha;
      acc[0][r] *= alpha; acc[1][r] *= alpha;
      acc[2][r] *= alpha; acc[3][r] *= alpha;
    }
#pragma unroll
    for (int nt = 0; nt < 4; ++nt)
#pragma unroll
      for (int r = 0; r < 8; ++r) sf[nt][r] = expf(sf[nt][r] - mnew[r]);
#pragma unroll
    for (int r = 0; r < 8; ++r) {
      float sm = sf[0][r] + sf[1][r] + sf[2][r] + sf[3][r];
#pragma unroll
      for (int off = 8; off >= 1; off >>= 1) sm += __shfl_xor(sm, off, 32);
      lrow[r] += sm;
    }

    // re-layout P (D-layout -> A-layout) via per-wave LDS scratch
#pragma unroll
    for (int nt = 0; nt < 4; ++nt)
#pragma unroll
      for (int r = 0; r < 8; ++r)
        Pw[(r + mhalf) * ALK + nt * 16 + ncol] = (bf16_t)sf[nt][r];

    // O += P * V  (V transposed in LDS so B frags are contiguous in keys)
#pragma unroll
    for (int c = 0; c < 2; ++c) {
      v16bf pf = load_a(Pw, ALK, c * 32, lane);
#pragma unroll
      for (int j = 0; j < 4; ++j)
        acc[j] = wmma_bf16f32(pf, load_b(Vt[cur], ALK, j * 16, c * 32, lane),
                              acc[j]);
    }

    async_wait0();   // next K tile landed in LDS
    __syncthreads(); // (also flushes the Vt[nxt] ds stores block-wide)
  }

  // normalize and write [b][n][h*64+d] bf16 for the projection GEMM
#pragma unroll
  for (int j = 0; j < 4; ++j)
#pragma unroll
    for (int r = 0; r < 8; ++r) {
      float o = acc[j][r] / lrow[r];
      int m = qBase + wave * 16 + r + mhalf;
      O[((size_t)b * S + m) * DIM + h * HD + j * 16 + ncol] = (bf16_t)o;
    }
}

// ---------------------------------------------------------------------------
// launch
// ---------------------------------------------------------------------------
extern "C" void kernel_launch(void* const* d_in, const int* in_sizes, int n_in,
                              void* d_out, int out_size, void* d_ws,
                              size_t ws_size, hipStream_t stream) {
  (void)in_sizes; (void)n_in; (void)out_size; (void)ws_size;
  const float* x       = (const float*)d_in[0];
  const float* qkv_w   = (const float*)d_in[1];
  const float* qkv_b   = (const float*)d_in[2];
  const float* q_scale = (const float*)d_in[3];
  const float* k_scale = (const float*)d_in[4];
  const float* proj_w  = (const float*)d_in[5];
  const float* proj_b  = (const float*)d_in[6];
  const int* heightp   = (const int*)d_in[7];
  const int* widthp    = (const int*)d_in[8];
  (void)heightp;

  const int B = 2, S = 2304, DIN = 1024, DIM = 1024;
  const int M = B * S, N3 = 3 * DIM;

  char* ws = (char*)d_ws;
  size_t off = 0;
  bf16_t* Xb  = (bf16_t*)(ws + off); off += (size_t)M * DIN * 2;
  bf16_t* Wq  = (bf16_t*)(ws + off); off += (size_t)N3 * DIN * 2;
  bf16_t* Wp  = (bf16_t*)(ws + off); off += (size_t)DIM * DIN * 2;
  float*  QKV = (float*)(ws + off);  off += (size_t)M * N3 * 4;
  bf16_t* Qb  = (bf16_t*)(ws + off); off += (size_t)M * DIM * 2;
  bf16_t* Kb  = (bf16_t*)(ws + off); off += (size_t)M * DIM * 2;
  bf16_t* Vb  = (bf16_t*)(ws + off); off += (size_t)M * DIM * 2;
  bf16_t* AO  = (bf16_t*)(ws + off); off += (size_t)M * DIM * 2;

  f32_to_bf16_kernel<<<1024, 256, 0, stream>>>(x, Xb, M * DIN);
  f32_to_bf16_kernel<<<1024, 256, 0, stream>>>(qkv_w, Wq, N3 * DIN);
  f32_to_bf16_kernel<<<1024, 256, 0, stream>>>(proj_w, Wp, DIM * DIN);

  gemm_bf16_nt<<<dim3(N3 / 128, M / 128), 256, 0, stream>>>(
      Xb, Wq, qkv_b, QKV, M, N3, DIN);

  norm_rope_kernel<<<dim3(S, B), 256, 0, stream>>>(
      QKV, q_scale, k_scale, Qb, Kb, Vb, widthp);

  attn_fwd_kernel<<<dim3(S / 128, B * 16), 256, 0, stream>>>(Qb, Kb, Vb, AO);

  gemm_bf16_nt<<<dim3(DIM / 128, M / 128), 256, 0, stream>>>(
      AO, Wp, proj_b, (float*)d_out, M, DIM, DIN);
}
